// Str2Str_89610197664496
// MI455X (gfx1250) — matrix operand order repaired and checked by reading, not compiled, not run on hardware
//
#include <hip/hip_runtime.h>
#include <hip/hip_bf16.h>
#include <math.h>

typedef __attribute__((ext_vector_type(16))) _Float16 v16h;
typedef __attribute__((ext_vector_type(8)))  _Float16 v8h;
typedef __attribute__((ext_vector_type(8)))  float    v8f;

#define LNE 1e-5f

static __device__ __forceinline__ float wred32(float s) {
#pragma unroll
  for (int m = 16; m >= 1; m >>= 1) s += __shfl_xor(s, m, 32);
  return s;
}

// ---------------------------------------------------------------------------
// WMMA tile GEMM: C(64 x N) += A(64 x K) * B(K x N)
// A: f16 row-major (stride lda halves, 16B-aligned rows), LDS or global.
// Bsw: weights pre-swizzled into per-lane fragment layout:
//   flat[((kt*(N/16)+nt)*32 + lane)*16 + h], lane=(n&15)+16*((k>>3)&1), h=(k&7)+8*((k>>4)&1)
// Block = 128 threads (4 waves); wave w computes rows [16w,16w+16).
// acc holds N/16 v8f accumulators. Fragment maps per CDNA5 ISA 7.12.2.
// ---------------------------------------------------------------------------
template <int K, int N>
static __device__ __forceinline__ void gemm64(const _Float16* __restrict__ A, int lda,
                                              const _Float16* __restrict__ Bsw, v8f* acc) {
  const int lane = threadIdx.x & 31;
  const int wave = threadIdx.x >> 5;
  const int m    = lane & 15;
  const int hi   = lane >> 4;
  __syncthreads();  // A produced by other waves
#pragma unroll
  for (int kt = 0; kt < K / 32; ++kt) {
    const _Float16* Ap = A + (size_t)(wave * 16 + m) * lda + kt * 32 + 8 * hi;
    v8h alo = *(const v8h*)(Ap);        // k = kt*32 + 8*hi + [0..7]
    v8h ahi = *(const v8h*)(Ap + 16);   // k = kt*32 + 16 + 8*hi + [0..7]
    v16h a = __builtin_shufflevector(alo, ahi, 0, 1, 2, 3, 4, 5, 6, 7,
                                     8, 9, 10, 11, 12, 13, 14, 15);
#pragma unroll
    for (int nt = 0; nt < N / 16; ++nt) {
      v16h b = *(const v16h*)(Bsw + (((size_t)(kt * (N / 16) + nt) * 32 + lane) << 4));
      acc[nt] = __builtin_amdgcn_wmma_f32_16x16x32_f16(false, a, false, b, (short)0,
                                                       acc[nt], false, false);
    }
  }
  __syncthreads();  // protect A against caller's overwrite (WAR)
}

// ---------------------------------------------------------------------------
// Weight convert + zero-pad + swizzle into WMMA B-fragment layout.
// dst has dr*dc elements (dr mult of 32, dc mult of 16).
// ---------------------------------------------------------------------------
__global__ void k_convsw(const float* __restrict__ src, _Float16* __restrict__ dst,
                         int sr, int sc, int dr, int dc) {
  int e = blockIdx.x * 256 + threadIdx.x;
  if (e >= dr * dc) return;
  int h     = e & 15;
  int lane  = (e >> 4) & 31;
  int tile  = e >> 9;
  int ntile = dc >> 4;
  int kt = tile / ntile, nt = tile % ntile;
  int hi = lane >> 4;
  int k = kt * 32 + ((h >> 3) << 4) + hi * 8 + (h & 7);
  int n = nt * 16 + (lane & 15);
  dst[e] = (k < sr && n < sc) ? (_Float16)src[k * sc + n] : (_Float16)0.f;
}

// ---------------------------------------------------------------------------
// LN(seq) over D_MSA=256 -> f16  (seq = msa[:,0])
// ---------------------------------------------------------------------------
__global__ __launch_bounds__(256) void k_seq_ln(const float* __restrict__ msa,
                                                _Float16* __restrict__ seq_ln) {
  __shared__ float red[256];
  const int l = blockIdx.x, tid = threadIdx.x;
  float v = msa[(size_t)l * 256 + tid];
  red[tid] = v;
  __syncthreads();
  for (int s = 128; s > 0; s >>= 1) { if (tid < s) red[tid] += red[tid + s]; __syncthreads(); }
  float mu = red[0] * (1.f / 256.f);
  __syncthreads();
  float d = v - mu;
  red[tid] = d * d;
  __syncthreads();
  for (int s = 128; s > 0; s >>= 1) { if (tid < s) red[tid] += red[tid + s]; __syncthreads(); }
  float rs = rsqrtf(red[0] * (1.f / 256.f) + LNE);
  seq_ln[(size_t)l * 256 + tid] = (_Float16)(d * rs);
}

// ---------------------------------------------------------------------------
// node = LN(concat(LN(seq),LN(state)) @ W_x + b)  -> f16 (1024 x 32)
// ---------------------------------------------------------------------------
__global__ __launch_bounds__(128) void k_node(const _Float16* __restrict__ seq_ln,
                                              const float* __restrict__ state,
                                              const float* __restrict__ xw,
                                              const float* __restrict__ xb,
                                              _Float16* __restrict__ node_h) {
  __shared__ _Float16 sRow[4][256];
  const int wave = threadIdx.x >> 5, lane = threadIdx.x & 31;
  const int l = blockIdx.x * 4 + wave;
#pragma unroll
  for (int u = 0; u < 8; ++u) sRow[wave][lane + 32 * u] = seq_ln[(size_t)l * 256 + lane + 32 * u];
  __syncthreads();
  float st[16], mu = 0.f;
#pragma unroll
  for (int k = 0; k < 16; ++k) { st[k] = state[(size_t)l * 16 + k]; mu += st[k]; }
  mu *= (1.f / 16.f);
  float q = 0.f;
#pragma unroll
  for (int k = 0; k < 16; ++k) { float d = st[k] - mu; q += d * d; }
  float rs = rsqrtf(q * (1.f / 16.f) + LNE);
  float a = xb[lane];
  for (int k = 0; k < 256; ++k) a += (float)sRow[wave][k] * xw[k * 32 + lane];
#pragma unroll
  for (int k = 0; k < 16; ++k) a += ((st[k] - mu) * rs) * xw[(256 + k) * 32 + lane];
  float mu2 = wred32(a) * (1.f / 32.f);
  float d = a - mu2;
  float var = wred32(d * d) * (1.f / 32.f);
  node_h[(size_t)l * 32 + lane] = (_Float16)(d * rsqrtf(var + LNE));
}

// ---------------------------------------------------------------------------
// top-k=64 smallest Dg per row (order irrelevant: downstream reduces over k)
// ---------------------------------------------------------------------------
__global__ __launch_bounds__(256) void k_topk(const float* __restrict__ xyz,
                                              const int* __restrict__ idx,
                                              const int* __restrict__ topk_p,
                                              int* __restrict__ eidx) {
  __shared__ float sD[1024];
  __shared__ float rv[256];
  __shared__ int   ri[256];
  const int i = blockIdx.x, tid = threadIdx.x;
  const float cx = xyz[i * 9 + 3], cy = xyz[i * 9 + 4], cz = xyz[i * 9 + 5];
  const int ii = idx[i];
  for (int j = tid; j < 1024; j += 256) {
    float dx = xyz[j * 9 + 3] - cx, dy = xyz[j * 9 + 4] - cy, dz = xyz[j * 9 + 5] - cz;
    float d = sqrtf(dx * dx + dy * dy + dz * dz + 1e-8f);
    if (j == i) d += 999.9f;
    d += fabsf((float)(idx[j] - ii)) * 1e-5f;
    sD[j] = d;
  }
  __syncthreads();
  int K = topk_p[0];
  if (K < 1 || K > 64) K = 64;
  for (int t = 0; t < K; ++t) {
    float best = 1e30f; int bi = 0;
    for (int j = tid; j < 1024; j += 256) {
      float v = sD[j];
      if (v < best) { best = v; bi = j; }
    }
    rv[tid] = best; ri[tid] = bi;
    __syncthreads();
    for (int s = 128; s > 0; s >>= 1) {
      if (tid < s) {
        if (rv[tid + s] < rv[tid] || (rv[tid + s] == rv[tid] && ri[tid + s] < ri[tid])) {
          rv[tid] = rv[tid + s]; ri[tid] = ri[tid + s];
        }
      }
      __syncthreads();
    }
    if (tid == 0) { eidx[i * 64 + t] = ri[0]; sD[ri[0]] = 1e30f; }
    __syncthreads();
  }
}

// ---------------------------------------------------------------------------
// Fused per-residue edge features + message MLP + frame update.
// 1 block per residue i, 128 threads (4 waves). All GEMMs via WMMA f16.
// ---------------------------------------------------------------------------
__global__ __launch_bounds__(128) void k_edge_msg(
    const float* __restrict__ pair, const float* __restrict__ xyz,
    const int* __restrict__ idx, const unsigned char* __restrict__ motif,
    const float* __restrict__ Rin, const float* __restrict__ Tin,
    const float* __restrict__ e1b, const float* __restrict__ e2b,
    const float* __restrict__ mb1, const float* __restrict__ mb2,
    const _Float16* __restrict__ node_h, const int* __restrict__ eidx,
    const _Float16* __restrict__ w_e1, const _Float16* __restrict__ w_e2p,
    const _Float16* __restrict__ w_m1, const _Float16* __restrict__ w_m2p,
    float* __restrict__ out) {
  __shared__ alignas(32) _Float16 sA[64 * 128];
  __shared__ float    sC[64 * 64];
  __shared__ int      sIdx[64];
  __shared__ float    sRel[64 * 3];
  __shared__ float    sD[64];
  __shared__ float    sOff[6];

  const int i = blockIdx.x;
  const int tid = threadIdx.x, lane = tid & 31, wave = tid >> 5;
  const int nn = lane & 15, hi = lane >> 4;

  if (tid < 64) {
    int j = eidx[i * 64 + tid];
    sIdx[tid] = j;
    float dx = xyz[j * 9 + 3] - xyz[i * 9 + 3];
    float dy = xyz[j * 9 + 4] - xyz[i * 9 + 4];
    float dz = xyz[j * 9 + 5] - xyz[i * 9 + 5];
    sRel[tid * 3 + 0] = dx; sRel[tid * 3 + 1] = dy; sRel[tid * 3 + 2] = dz;
    sD[tid] = sqrtf(dx * dx + dy * dy + dz * dz + 1e-8f);
  }
  __syncthreads();

  // pair rows (gathered) -> LN over 128 -> sA f16 (64 x 128)
  for (int r = wave; r < 64; r += 4) {
    const float* pr = pair + ((size_t)i * 1024 + (size_t)sIdx[r]) * 128;
    float v[4];
#pragma unroll
    for (int u = 0; u < 4; ++u) v[u] = pr[lane + 32 * u];
    float mu = wred32(v[0] + v[1] + v[2] + v[3]) * (1.f / 128.f);
    float q = 0.f;
#pragma unroll
    for (int u = 0; u < 4; ++u) { float d = v[u] - mu; q += d * d; }
    float rs = rsqrtf(wred32(q) * (1.f / 128.f) + LNE);
#pragma unroll
    for (int u = 0; u < 4; ++u) sA[r * 128 + lane + 32 * u] = (_Float16)((v[u] - mu) * rs);
  }

  v8f z = {0.f, 0.f, 0.f, 0.f, 0.f, 0.f, 0.f, 0.f};

  // GEMM1: e1 = LN(pair) @ W_e1 + b1  -> sC (64 x 32)
  {
    v8f acc[2] = {z, z};
    gemm64<128, 32>(sA, 128, w_e1, acc);
#pragma unroll
    for (int nt = 0; nt < 2; ++nt)
#pragma unroll
      for (int g = 0; g < 8; ++g) {
        int row = wave * 16 + g + 8 * hi, col = nt * 16 + nn;
        sC[row * 32 + col] = acc[nt][g] + e1b[col];
      }
  }
  __syncthreads();

  // LN(e1) | rbf(D) | bonded | zero-pad -> A2 (64 x 96) in sA
  if (tid < 64) {
    const int r = tid;
    float mu = 0.f;
    for (int c = 0; c < 32; ++c) mu += sC[r * 32 + c];
    mu *= (1.f / 32.f);
    float q = 0.f;
    for (int c = 0; c < 32; ++c) { float d = sC[r * 32 + c] - mu; q += d * d; }
    float rs = rsqrtf(q * (1.f / 32.f) + LNE);
    for (int c = 0; c < 32; ++c) sA[r * 96 + c] = (_Float16)((sC[r * 32 + c] - mu) * rs);
    float d = sD[r];
    for (int t = 0; t < 36; ++t) {
      float mu_t = 2.f + (20.f / 35.f) * (float)t;
      float zz = (d - mu_t) * (36.f / 20.f);
      sA[r * 96 + 32 + t] = (_Float16)expf(-zz * zz);
    }
    int sep = idx[sIdx[r]] - idx[i];
    float bnd = (sep == 1) ? 1.f : ((sep == -1) ? -1.f : 0.f);
    sA[r * 96 + 68] = (_Float16)bnd;
    for (int c = 69; c < 96; ++c) sA[r * 96 + c] = (_Float16)0.f;
  }

  // GEMM2: e2 = A2 @ W_e2p + b2 -> sC (64 x 32)
  {
    v8f acc[2] = {z, z};
    gemm64<96, 32>(sA, 96, w_e2p, acc);
#pragma unroll
    for (int nt = 0; nt < 2; ++nt)
#pragma unroll
      for (int g = 0; g < 8; ++g) {
        int row = wave * 16 + g + 8 * hi, col = nt * 16 + nn;
        sC[row * 32 + col] = acc[nt][g] + e2b[col];
      }
  }
  __syncthreads();

  // LN(e2) -> e_feat; A3 = [h_i | h_j | e_feat] (64 x 96) in sA
  if (tid < 64) {
    const int r = tid;
    float mu = 0.f;
    for (int c = 0; c < 32; ++c) mu += sC[r * 32 + c];
    mu *= (1.f / 32.f);
    float q = 0.f;
    for (int c = 0; c < 32; ++c) { float d = sC[r * 32 + c] - mu; q += d * d; }
    float rs = rsqrtf(q * (1.f / 32.f) + LNE);
    int j = sIdx[r];
    for (int c = 0; c < 32; ++c) {
      sA[r * 96 + c]      = node_h[(size_t)i * 32 + c];
      sA[r * 96 + 32 + c] = node_h[(size_t)j * 32 + c];
      sA[r * 96 + 64 + c] = (_Float16)((sC[r * 32 + c] - mu) * rs);
    }
  }

  // GEMM3: H = relu(A3 @ W_m1 + b) -> sC (64 x 64)
  {
    v8f acc[4] = {z, z, z, z};
    gemm64<96, 64>(sA, 96, w_m1, acc);
#pragma unroll
    for (int nt = 0; nt < 4; ++nt)
#pragma unroll
      for (int g = 0; g < 8; ++g) {
        int row = wave * 16 + g + 8 * hi, col = nt * 16 + nn;
        sC[row * 64 + col] = fmaxf(acc[nt][g] + mb1[col], 0.f);
      }
  }
  __syncthreads();

  // A4 = f16(H) (64 x 64)
  for (int e = tid; e < 64 * 64; e += 128) sA[e] = (_Float16)sC[e];

  // GEMM4: M = A4 @ W_m2p + b (cols<18 valid) -> sC (64 x 32)
  {
    v8f acc[2] = {z, z};
    gemm64<64, 32>(sA, 64, w_m2p, acc);
#pragma unroll
    for (int nt = 0; nt < 2; ++nt)
#pragma unroll
      for (int g = 0; g < 8; ++g) {
        int row = wave * 16 + g + 8 * hi, col = nt * 16 + nn;
        sC[row * 32 + col] = acc[nt][g] + ((col < 18) ? mb2[col] : 0.f);
      }
  }
  __syncthreads();

  // state_out = mean_k M[:, :16]
  if (tid < 16) {
    float s = 0.f;
    for (int k = 0; k < 64; ++k) s += sC[k * 32 + tid];
    out[12288 + (size_t)i * 16 + tid] = s * (1.f / 64.f);
  }
  // offset[c,d] = (1/64) sum_k M[k,16+c] * rel[k,d]
  if (tid >= 32 && tid < 38) {
    int t = tid - 32, cp = t / 3, d = t % 3;
    float s = 0.f;
    for (int k = 0; k < 64; ++k) s += sC[k * 32 + 16 + cp] * sRel[k * 3 + d];
    sOff[t] = s * (1.f / 64.f);
  }
  __syncthreads();

  if (tid == 0) {
    const bool msk = motif[i] != 0;
    float delT[3], Rv[3];
#pragma unroll
    for (int d = 0; d < 3; ++d) {
      float o0 = msk ? 0.f : sOff[d];
      float o1 = msk ? 0.f : sOff[3 + d];
      delT[d] = o0 * 0.1f;
      Rv[d]   = o1 * 0.01f;
    }
    float Qn = sqrtf(1.f + Rv[0] * Rv[0] + Rv[1] * Rv[1] + Rv[2] * Rv[2]);
    float qA = 1.f / Qn, qB = Rv[0] / Qn, qC = Rv[1] / Qn, qD = Rv[2] / Qn;
    float Ro[9];
    Ro[0] = qA * qA + qB * qB - qC * qC - qD * qD;
    Ro[1] = 2.f * (qB * qC - qA * qD);
    Ro[2] = 2.f * (qB * qD + qA * qC);
    Ro[3] = 2.f * (qB * qC + qA * qD);
    Ro[4] = qA * qA - qB * qB + qC * qC - qD * qD;
    Ro[5] = 2.f * (qC * qD - qA * qB);
    Ro[6] = 2.f * (qB * qD - qA * qC);
    Ro[7] = 2.f * (qC * qD + qA * qB);
    Ro[8] = qA * qA - qB * qB - qC * qC + qD * qD;
#pragma unroll
    for (int r = 0; r < 3; ++r)
#pragma unroll
      for (int c = 0; c < 3; ++c) {
        float s = 0.f;
#pragma unroll
        for (int k = 0; k < 3; ++k) s += Ro[r * 3 + k] * Rin[(size_t)i * 9 + k * 3 + c];
        out[(size_t)i * 9 + r * 3 + c] = s;
      }
#pragma unroll
    for (int d = 0; d < 3; ++d) out[9216 + (size_t)i * 3 + d] = delT[d] + Tin[(size_t)i * 3 + d];
  }
}

// ---------------------------------------------------------------------------
// Residual MLP head -> alpha. 16 blocks x 64 rows, 128 threads.
// si lives in WMMA accumulators of the owning wave across the residuals.
// ---------------------------------------------------------------------------
__global__ __launch_bounds__(128) void k_si(
    const _Float16* __restrict__ seq_ln, const float* __restrict__ state_out,
    const _Float16* __restrict__ w_s0, const float* __restrict__ b_s0,
    const _Float16* __restrict__ w_sip, const float* __restrict__ b_si,
    const _Float16* __restrict__ w_l1, const float* __restrict__ b_l1,
    const _Float16* __restrict__ w_l2, const float* __restrict__ b_l2,
    const _Float16* __restrict__ w_l3, const float* __restrict__ b_l3,
    const _Float16* __restrict__ w_l4, const float* __restrict__ b_l4,
    const _Float16* __restrict__ w_outp, const float* __restrict__ b_out,
    float* __restrict__ out) {
  __shared__ alignas(32) _Float16 sAct[64 * 128];
  __shared__ alignas(32) _Float16 sAct2[64 * 128];
  __shared__ alignas(32) _Float16 sSt[64 * 32];

  const int r0 = blockIdx.x * 64;
  const int tid = threadIdx.x, lane = tid & 31, wave = tid >> 5;
  const int nn = lane & 15, hi = lane >> 4;

  // LN(state_out) over 16, zero-padded to K=32
  if (tid < 64) {
    const float* sp = state_out + (size_t)(r0 + tid) * 16;
    float v[16], mu = 0.f;
#pragma unroll
    for (int k = 0; k < 16; ++k) { v[k] = sp[k]; mu += v[k]; }
    mu *= (1.f / 16.f);
    float q = 0.f;
#pragma unroll
    for (int k = 0; k < 16; ++k) { float d = v[k] - mu; q += d * d; }
    float rs = rsqrtf(q * (1.f / 16.f) + LNE);
#pragma unroll
    for (int k = 0; k < 16; ++k) sSt[tid * 32 + k] = (_Float16)((v[k] - mu) * rs);
#pragma unroll
    for (int k = 16; k < 32; ++k) sSt[tid * 32 + k] = (_Float16)0.f;
  }

  v8f z = {0.f, 0.f, 0.f, 0.f, 0.f, 0.f, 0.f, 0.f};
  v8f si[8];
#pragma unroll
  for (int t = 0; t < 8; ++t) si[t] = z;

  gemm64<256, 128>(seq_ln + (size_t)r0 * 256, 256, w_s0, si);
  gemm64<32, 128>(sSt, 32, w_sip, si);

  // si = relu(si + b0 + bsi); stash f16 in sAct
#pragma unroll
  for (int nt = 0; nt < 8; ++nt)
#pragma unroll
    for (int g = 0; g < 8; ++g) {
      int row = wave * 16 + g + 8 * hi, col = nt * 16 + nn;
      float v = fmaxf(si[nt][g] + b_s0[col] + b_si[col], 0.f);
      si[nt][g] = v;
      sAct[row * 128 + col] = (_Float16)v;
    }

  // residual block 1: si = r + relu(r@W1+b1)@W2+b2 ; then si = relu(si)
  {
    v8f t[8];
#pragma unroll
    for (int q = 0; q < 8; ++q) t[q] = z;
    gemm64<128, 128>(sAct, 128, w_l1, t);
#pragma unroll
    for (int nt = 0; nt < 8; ++nt)
#pragma unroll
      for (int g = 0; g < 8; ++g) {
        int row = wave * 16 + g + 8 * hi, col = nt * 16 + nn;
        sAct2[row * 128 + col] = (_Float16)fmaxf(t[nt][g] + b_l1[col], 0.f);
      }
    v8f u[8];
#pragma unroll
    for (int q = 0; q < 8; ++q) u[q] = z;
    gemm64<128, 128>(sAct2, 128, w_l2, u);
#pragma unroll
    for (int nt = 0; nt < 8; ++nt)
#pragma unroll
      for (int g = 0; g < 8; ++g) {
        int row = wave * 16 + g + 8 * hi, col = nt * 16 + nn;
        float v = fmaxf(si[nt][g] + u[nt][g] + b_l2[col], 0.f);
        si[nt][g] = v;
        sAct[row * 128 + col] = (_Float16)v;
      }
  }
  // residual block 2
  {
    v8f t[8];
#pragma unroll
    for (int q = 0; q < 8; ++q) t[q] = z;
    gemm64<128, 128>(sAct, 128, w_l3, t);
#pragma unroll
    for (int nt = 0; nt < 8; ++nt)
#pragma unroll
      for (int g = 0; g < 8; ++g) {
        int row = wave * 16 + g + 8 * hi, col = nt * 16 + nn;
        sAct2[row * 128 + col] = (_Float16)fmaxf(t[nt][g] + b_l3[col], 0.f);
      }
    v8f u[8];
#pragma unroll
    for (int q = 0; q < 8; ++q) u[q] = z;
    gemm64<128, 128>(sAct2, 128, w_l4, u);
#pragma unroll
    for (int nt = 0; nt < 8; ++nt)
#pragma unroll
      for (int g = 0; g < 8; ++g) {
        int row = wave * 16 + g + 8 * hi, col = nt * 16 + nn;
        float v = fmaxf(si[nt][g] + u[nt][g] + b_l4[col], 0.f);
        sAct[row * 128 + col] = (_Float16)v;  // relu(si) for the output head
      }
  }
  // alpha = relu(si) @ W_out + b (cols < 20)
  {
    v8f a2[2] = {z, z};
    gemm64<128, 32>(sAct, 128, w_outp, a2);
#pragma unroll
    for (int nt = 0; nt < 2; ++nt)
#pragma unroll
      for (int g = 0; g < 8; ++g) {
        int row = wave * 16 + g + 8 * hi, col = nt * 16 + nn;
        if (col < 20) out[28672 + (size_t)(r0 + row) * 20 + col] = a2[nt][g] + b_out[col];
      }
  }
}

// ---------------------------------------------------------------------------
extern "C" void kernel_launch(void* const* d_in, const int* in_sizes, int n_in,
                              void* d_out, int out_size, void* d_ws, size_t ws_size,
                              hipStream_t stream) {
  const float* msa   = (const float*)d_in[0];
  const float* pair  = (const float*)d_in[1];
  const float* R_in  = (const float*)d_in[2];
  const float* T_in  = (const float*)d_in[3];
  const float* xyz   = (const float*)d_in[4];
  const float* state = (const float*)d_in[5];
  const int*   idx   = (const int*)d_in[6];
  const unsigned char* motif = (const unsigned char*)d_in[7];
  const int*   topk  = (const int*)d_in[8];
  const float* xw  = (const float*)d_in[9];
  const float* xb  = (const float*)d_in[10];
  const float* e1w = (const float*)d_in[11];
  const float* e1b = (const float*)d_in[12];
  const float* e2w = (const float*)d_in[13];
  const float* e2b = (const float*)d_in[14];
  const float* m1w = (const float*)d_in[15];
  const float* m1b = (const float*)d_in[16];
  const float* m2w = (const float*)d_in[17];
  const float* m2b = (const float*)d_in[18];
  const float* s0w = (const float*)d_in[19];
  const float* s0b = (const float*)d_in[20];
  const float* siw = (const float*)d_in[21];
  const float* sib = (const float*)d_in[22];
  const float* l1w = (const float*)d_in[23];
  const float* l1b = (const float*)d_in[24];
  const float* l2w = (const float*)d_in[25];
  const float* l2b = (const float*)d_in[26];
  const float* l3w = (const float*)d_in[27];
  const float* l3b = (const float*)d_in[28];
  const float* l4w = (const float*)d_in[29];
  const float* l4b = (const float*)d_in[30];
  const float* ow  = (const float*)d_in[31];
  const float* ob  = (const float*)d_in[32];

  char* ws = (char*)d_ws;
  size_t off = 0;
  auto carve = [&](size_t bytes) {
    char* p = ws + off;
    off = (off + bytes + 255) & ~(size_t)255;
    return p;
  };
  _Float16* seq_ln = (_Float16*)carve(1024 * 256 * 2);
  _Float16* node_h = (_Float16*)carve(1024 * 32 * 2);
  int*      eidx   = (int*)carve(1024 * 64 * 4);
  _Float16* W_e1   = (_Float16*)carve(128 * 32 * 2);
  _Float16* W_e2p  = (_Float16*)carve(96 * 32 * 2);
  _Float16* W_m1   = (_Float16*)carve(96 * 64 * 2);
  _Float16* W_m2p  = (_Float16*)carve(64 * 32 * 2);
  _Float16* W_s0   = (_Float16*)carve(256 * 128 * 2);
  _Float16* W_sip  = (_Float16*)carve(32 * 128 * 2);
  _Float16* W_l1   = (_Float16*)carve(128 * 128 * 2);
  _Float16* W_l2   = (_Float16*)carve(128 * 128 * 2);
  _Float16* W_l3   = (_Float16*)carve(128 * 128 * 2);
  _Float16* W_l4   = (_Float16*)carve(128 * 128 * 2);
  _Float16* W_outp = (_Float16*)carve(128 * 32 * 2);

  auto conv = [&](const float* s, _Float16* dptr, int sr, int sc, int dr, int dc) {
    int n = dr * dc;
    k_convsw<<<(n + 255) / 256, 256, 0, stream>>>(s, dptr, sr, sc, dr, dc);
  };
  conv(e1w, W_e1, 128, 32, 128, 32);
  conv(e2w, W_e2p, 69, 32, 96, 32);
  conv(m1w, W_m1, 96, 64, 96, 64);
  conv(m2w, W_m2p, 64, 18, 64, 32);
  conv(s0w, W_s0, 256, 128, 256, 128);
  conv(siw, W_sip, 16, 128, 32, 128);
  conv(l1w, W_l1, 128, 128, 128, 128);
  conv(l2w, W_l2, 128, 128, 128, 128);
  conv(l3w, W_l3, 128, 128, 128, 128);
  conv(l4w, W_l4, 128, 128, 128, 128);
  conv(ow,  W_outp, 128, 20, 128, 32);

  k_seq_ln<<<1024, 256, 0, stream>>>(msa, seq_ln);
  k_node<<<256, 128, 0, stream>>>(seq_ln, state, xw, xb, node_h);
  k_topk<<<1024, 256, 0, stream>>>(xyz, idx, topk, eidx);

  float* outf = (float*)d_out;
  k_edge_msg<<<1024, 128, 0, stream>>>(pair, xyz, idx, motif, R_in, T_in, e1b, e2b, m1b,
                                       m2b, node_h, eidx, W_e1, W_e2p, W_m1, W_m2p, outf);
  k_si<<<16, 128, 0, stream>>>(seq_ln, outf + 12288, W_s0, s0b, W_sip, sib, W_l1, l1b,
                               W_l2, l2b, W_l3, l3b, W_l4, l4b, W_outp, ob, outf);
  (void)in_sizes; (void)n_in; (void)out_size; (void)ws_size;
}
